// AnimeInterp_13649406066889
// MI455X (gfx1250) — matrix operand order repaired and checked by reading, not compiled, not run on hardware
//
#include <hip/hip_runtime.h>
#include <hip/hip_fp16.h>

// ---------------------------------------------------------------------------
// AnimeInterp forward pipeline for gfx1250 (MI455X).
// Convs run as implicit GEMM on v_wmma_f32_16x16x32_f16 (f16 in, f32 acc),
// with pre-packed A tiles and branchless clamped im2col gathers.
// ---------------------------------------------------------------------------

typedef __attribute__((ext_vector_type(16))) _Float16 v16h;
typedef __attribute__((ext_vector_type(8)))  float    v8f;

#define THREADS 256
#define PT 4   // pixel tiles (16 px each) per wave

static inline int nblk(long n, int t) { return (int)((n + t - 1) / t); }

// ---------------------------------------------------------------------------
// Elementwise kernels
// ---------------------------------------------------------------------------
__global__ void zero_f32(float* __restrict__ p, long n) {
    long i = (long)blockIdx.x * blockDim.x + threadIdx.x;
    if (i < n) p[i] = 0.f;
}

__global__ void add_f32(float* __restrict__ dst, const float* __restrict__ src, long n) {
    long i = (long)blockIdx.x * blockDim.x + threadIdx.x;
    if (i < n) dst[i] += src[i];
}

// out[i] = in[i] * (a + b * (*t))
__global__ void scale_flow(float* __restrict__ out, const float* __restrict__ in, long n,
                           const float* __restrict__ t, float a, float b) {
    long i = (long)blockIdx.x * blockDim.x + threadIdx.x;
    if (i >= n) return;
    float f = a + b * t[0];
    out[i] = in[i] * f;
}

// ---------------------------------------------------------------------------
// Forward warp (softmax splatting): scatter pass + normalize pass
// ---------------------------------------------------------------------------
__global__ void splat_scatter(const float* __restrict__ x, const float* __restrict__ flow,
                              const float* __restrict__ Z, float* __restrict__ num,
                              float* __restrict__ den, int C, int h, int w) {
    int p = blockIdx.x * blockDim.x + threadIdx.x;
    int hw = h * w;
    if (p >= hw) return;
    int y = p / w, xx = p - y * w;
    float m  = __expf(Z[p]);
    float fx = (float)xx + flow[p];
    float fy = (float)y  + flow[hw + p];
    int x0 = (int)floorf(fx), y0 = (int)floorf(fy);
    float tx = fx - (float)x0, ty = fy - (float)y0;
    float wg[4] = {(1.f - tx) * (1.f - ty), tx * (1.f - ty), (1.f - tx) * ty, tx * ty};
    int xs[4] = {x0, x0 + 1, x0, x0 + 1};
    int ys[4] = {y0, y0, y0 + 1, y0 + 1};
    for (int i = 0; i < 4; ++i) {
        int xi = xs[i], yi = ys[i];
        if (xi < 0 || xi >= w || yi < 0 || yi >= h) continue;
        float mw = m * wg[i];
        int idx = yi * w + xi;
        atomicAdd(den + idx, mw);
        for (int c = 0; c < C; ++c)
            atomicAdd(num + (long)c * hw + idx, x[(long)c * hw + p] * mw);
    }
}

__global__ void splat_norm(const float* __restrict__ num, const float* __restrict__ den,
                           float* __restrict__ out, int C, int hw) {
    long i = (long)blockIdx.x * blockDim.x + threadIdx.x;
    long n = (long)C * hw;
    if (i >= n) return;
    int p = (int)(i % hw);
    float d = den[p];
    float v = num[i];
    out[i] = (d > 0.f) ? (v / d) : v;
}

// ---------------------------------------------------------------------------
// Bilinear resize (matches reference: src = max((o+0.5)*scale - 0.5, 0))
// ---------------------------------------------------------------------------
__global__ void resize_bilinear_k(const float* __restrict__ in, float* __restrict__ out,
                                  int C, int h, int w, int oh, int ow, float postscale) {
    long i = (long)blockIdx.x * blockDim.x + threadIdx.x;
    long n = (long)C * oh * ow;
    if (i >= n) return;
    int ox = (int)(i % ow);
    long t1 = i / ow;
    int oy = (int)(t1 % oh);
    int c  = (int)(t1 / oh);
    float sy = fmaxf(((float)oy + 0.5f) * ((float)h / (float)oh) - 0.5f, 0.f);
    float sx = fmaxf(((float)ox + 0.5f) * ((float)w / (float)ow) - 0.5f, 0.f);
    int y0 = (int)floorf(sy), x0 = (int)floorf(sx);
    float wy = sy - (float)y0, wx = sx - (float)x0;
    int y1 = min(y0 + 1, h - 1), x1 = min(x0 + 1, w - 1);
    const float* ip = in + (long)c * h * w;
    float v00 = ip[y0 * w + x0], v01 = ip[y0 * w + x1];
    float v10 = ip[y1 * w + x0], v11 = ip[y1 * w + x1];
    float top = v00 * (1.f - wx) + v01 * wx;
    float bot = v10 * (1.f - wx) + v11 * wx;
    out[i] = (top * (1.f - wy) + bot * wy) * postscale;
}

// ---------------------------------------------------------------------------
// Pixel shuffle r=2: in (c4*4, h, w) -> out (c4, 2h, 2w)
// ---------------------------------------------------------------------------
__global__ void pixel_shuffle2(const float* __restrict__ in, float* __restrict__ out,
                               int c4, int h, int w) {
    long i = (long)blockIdx.x * blockDim.x + threadIdx.x;
    int ow = 2 * w, oh = 2 * h;
    long n = (long)c4 * oh * ow;
    if (i >= n) return;
    int ox = (int)(i % ow);
    long t1 = i / ow;
    int oy = (int)(t1 % oh);
    int c  = (int)(t1 / oh);
    int rx = ox & 1, ry = oy & 1;
    int x = ox >> 1, y = oy >> 1;
    out[i] = in[((c * 4 + ry * 2 + rx) * h + y) * w + x];
}

// ---------------------------------------------------------------------------
// Weight packer: OIHW fp32 -> f16 A-tiles in the ISA's 16-bit 16x32 A layout.
// packA[((kc*coutTiles + ct)*32 + lane)*16 + e]:
//   lanes 0-15:  elems 0-7 -> K 0-7,  elems 8-15 -> K 16-23   (K rel. to chunk)
//   lanes16-31:  elems 0-7 -> K 8-15, elems 8-15 -> K 24-31
// Out-of-range K / cout are zero-filled, so the conv loop needs no guards.
// ---------------------------------------------------------------------------
__global__ void pack_weights(const float* __restrict__ wgt, _Float16* __restrict__ packA,
                             int Cin, int Cout, int chunks) {
    int coutTiles = (Cout + 15) >> 4;
    long total = (long)chunks * coutTiles * 512;
    long i = (long)blockIdx.x * blockDim.x + threadIdx.x;
    if (i >= total) return;
    int e    = (int)(i & 15);
    int lane = (int)((i >> 4) & 31);
    long t   = i >> 9;
    int ct = (int)(t % coutTiles);
    int kc = (int)(t / coutTiles);
    int hl = lane >> 4, lm = lane & 15;
    int kA = kc * 32 + e + ((e >= 8) ? 8 : 0) + (hl ? 8 : 0);
    int co = ct * 16 + lm;
    float v = 0.f;
    int Ktot = Cin * 9;
    if (kA < Ktot && co < Cout) {
        int cin = kA / 9;
        int kid = kA - cin * 9;
        v = wgt[(co * Cin + cin) * 9 + kid];
    }
    packA[i] = (_Float16)v;
}

// ---------------------------------------------------------------------------
// 3x3 conv, pad=1, implicit GEMM on WMMA.
// One wave computes a 16(cout) x 64(pixels) strip: 4 accumulators sharing one
// packed A tile per K-chunk; 4 back-to-back v_wmma per chunk. All gathers are
// branchless (clamped address + cndmask-zero), so loads cluster and EXEC stays
// full everywhere.
// ---------------------------------------------------------------------------
__global__ void conv3x3_wmma(const float* __restrict__ in, const _Float16* __restrict__ packA,
                             const float* __restrict__ bias, const float* __restrict__ alpha,
                             float* __restrict__ out,
                             int Cin, int Hi, int Wi, int Cout, int Ho, int Wo,
                             int stride, int accumulate, int clamp01) {
    int lane = threadIdx.x & 31;
    int wave = threadIdx.x >> 5;
    int tile = blockIdx.x * (blockDim.x >> 5) + wave;
    int coutTiles = (Cout + 15) >> 4;
    int npix = Ho * Wo;
    int pixTiles = (npix + 15) >> 4;
    int ptGroups = (pixTiles + PT - 1) / PT;
    if (tile >= coutTiles * ptGroups) return;

    int ct  = tile % coutTiles;
    int ptg = tile / coutTiles;
    int hl  = lane >> 4;
    int lm  = lane & 15;
    int hl16 = hl << 4;
    int hw = Hi * Wi;

    int p0 = ptg * (PT * 16) + lm;

    // Per-tile pixel coords, pre-strided (yi = pys + ky, xi = pxs + kx)
    int  pys[PT], pxs[PT];
    bool pok[PT];
    #pragma unroll
    for (int j = 0; j < PT; ++j) {
        int pj = p0 + 16 * j;
        pok[j] = pj < npix;
        int pc = pok[j] ? pj : 0;
        int py = pc / Wo;
        int px = pc - py * Wo;
        pys[j] = py * stride - 1;
        pxs[j] = px * stride - 1;
    }

    bool useP = (alpha != nullptr);
    float av = useP ? alpha[0] : 0.f;

    int Ktot = Cin * 9;
    int chunks = (Ktot + 31) >> 5;

    const v16h* pA = (const v16h*)packA;

    v8f acc[PT];
    #pragma unroll
    for (int j = 0; j < PT; ++j) acc[j] = (v8f){};

    for (int kc = 0; kc < chunks; ++kc) {
        v16h A = pA[(kc * coutTiles + ct) * 32 + lane];  // 32B contiguous load

        v16h Bm[PT];
        #pragma unroll
        for (int e = 0; e < 16; ++e) {
            int kB = kc * 32 + e + ((e >= 8) ? 8 : 0) + (hl ? 8 : 0);
            bool kOK = kB < Ktot;
            int kS = kOK ? kB : 0;
            int cin = kS / 9;
            int kid = kS - cin * 9;
            int ky = kid / 3;
            int kx = kid - ky * 3;
            int cbase = cin * hw;
            #pragma unroll
            for (int j = 0; j < PT; ++j) {
                int yi = pys[j] + ky;
                int xi = pxs[j] + kx;
                bool ok = kOK && pok[j] && (yi >= 0) && (yi < Hi) && (xi >= 0) && (xi < Wi);
                int yc = min(max(yi, 0), Hi - 1);
                int xc = min(max(xi, 0), Wi - 1);
                float v = in[cbase + yc * Wi + xc];   // unconditional, clamped
                v = ok ? v : 0.f;
                if (useP) v = (v > 0.f) ? v : av * v;
                Bm[j][e] = (_Float16)v;
            }
        }

        #pragma unroll
        for (int j = 0; j < PT; ++j)
            acc[j] = __builtin_amdgcn_wmma_f32_16x16x32_f16(false, A, false, Bm[j],
                                                            (short)0, acc[j], false, false);
    }

    // Epilogue. C layout: lane%16 = column(pixel); VGPR r -> row r (+8 hi half)
    int coutBase = ct * 16;
    #pragma unroll
    for (int j = 0; j < PT; ++j) {
        int p = p0 + 16 * j;
        if (p < npix) {
            #pragma unroll
            for (int r = 0; r < 8; ++r) {
                int co2 = coutBase + r + (hl ? 8 : 0);
                if (co2 < Cout) {
                    long idx = (long)co2 * npix + p;
                    float v = acc[j][r] + bias[co2];
                    if (accumulate) v += out[idx];
                    if (clamp01) v = fminf(fmaxf(v, 0.f), 1.f);
                    out[idx] = v;
                }
            }
        }
    }
}

// ---------------------------------------------------------------------------
// Host-side helpers
// ---------------------------------------------------------------------------
struct Blk { const float *w1, *b1, *w2, *b2, *a1, *a2; };

static void launch_conv(hipStream_t s, const float* in, int Cin, int Hi, int Wi,
                        const float* w, const float* b, const float* alpha,
                        float* out, int Cout, int stride, int acc, int clamp,
                        _Float16* packbuf) {
    int Ho = (Hi - 1) / stride + 1;
    int Wo = (Wi - 1) / stride + 1;
    int npix = Ho * Wo;
    int chunks = (Cin * 9 + 31) / 32;
    int coutTiles = (Cout + 15) / 16;
    long pn = (long)chunks * coutTiles * 512;
    pack_weights<<<nblk(pn, THREADS), THREADS, 0, s>>>(w, packbuf, Cin, Cout, chunks);

    int ptGroups = (((npix + 15) / 16) + PT - 1) / PT;
    int tiles = coutTiles * ptGroups;
    int blocks = (tiles + 7) / 8;   // 8 wave32 per 256-thread block
    conv3x3_wmma<<<blocks, THREADS, 0, s>>>(in, packbuf, b, alpha, out,
                                            Cin, Hi, Wi, Cout, Ho, Wo, stride, acc, clamp);
}

static void run_res(hipStream_t s, const float* in, int Cin, int Hq, int Wq,
                    const Blk& b, int Cout, float* tmp, float* out, int acc, _Float16* pk) {
    launch_conv(s, in,  Cin,  Hq, Wq, b.w1, b.b1, b.a1, tmp, Cout, 1, 0, 0, pk);
    launch_conv(s, tmp, Cout, Hq, Wq, b.w2, b.b2, b.a2, out, Cout, 1, acc, 0, pk);
}

static void run_down(hipStream_t s, const float* in, int Cin, int Hq, int Wq,
                     const Blk& b, int Cout, float* tmp, float* out, int acc, _Float16* pk) {
    launch_conv(s, in,  Cin,  Hq, Wq, b.w1, b.b1, b.a1, tmp, Cout, 2, 0, 0, pk);
    launch_conv(s, tmp, Cout, Hq / 2, Wq / 2, b.w2, b.b2, b.a2, out, Cout, 1, acc, 0, pk);
}

static void run_up(hipStream_t s, const float* in, int Cin, int Hq, int Wq,
                   const Blk& b, int Cout, float* rbuf, float* tmp, float* out, int acc,
                   _Float16* pk) {
    int oh = 2 * Hq, ow = 2 * Wq;
    long n = (long)Cin * oh * ow;
    resize_bilinear_k<<<nblk(n, THREADS), THREADS, 0, s>>>(in, rbuf, Cin, Hq, Wq, oh, ow, 1.f);
    launch_conv(s, rbuf, Cin,  oh, ow, b.w1, b.b1, b.a1, tmp, Cout, 1, 0, 0, pk);
    launch_conv(s, tmp,  Cout, oh, ow, b.w2, b.b2, b.a2, out, Cout, 1, acc, 0, pk);
}

static void run_warp_norm(hipStream_t s, const float* x, int C, const float* flow,
                          const float* Z, int h, int w, float* num, float* den, float* out) {
    long hw = (long)h * w;
    zero_f32<<<nblk(C * hw, THREADS), THREADS, 0, s>>>(num, C * hw);
    zero_f32<<<nblk(hw, THREADS), THREADS, 0, s>>>(den, hw);
    splat_scatter<<<nblk(hw, THREADS), THREADS, 0, s>>>(x, flow, Z, num, den, C, h, w);
    splat_norm<<<nblk(C * hw, THREADS), THREADS, 0, s>>>(num, den, out, C, (int)hw);
}

// ---------------------------------------------------------------------------
// Pipeline driver.
// Input flattening (setup_inputs dict insertion order, params recursively):
//  0:I1 1:I2 2:F12 3:F21 4:Z1 5:Z2 6:feat11 7:feat12 8:feat13 9:feat21
// 10:feat22 11:feat23 12:merged 13:t
// 14..145: 22 blocks (sorted: d10,d11,d20,d21,head,head1..head4,r01,r04,r05,
//          r11,r14,r15,r21,r24,r25,u04,u05,u14,u15) x {w1,b1,w2,b2,a1,a2}
// 146..152: tail {w1,b1,w2,b2,w3,b3,a1}
// ---------------------------------------------------------------------------
extern "C" void kernel_launch(void* const* d_in, const int* in_sizes, int n_in,
                              void* d_out, int out_size, void* d_ws, size_t ws_size,
                              hipStream_t stream) {
    (void)in_sizes; (void)n_in; (void)out_size; (void)ws_size;

    const int H = 448, W = 768, H2 = 224, W2 = 384, H4 = 112, W4 = 192;
    const long PX = (long)H * W, PX2 = (long)H2 * W2, PX4 = (long)H4 * W4;

    const float* I1     = (const float*)d_in[0];
    const float* I2     = (const float*)d_in[1];
    const float* F12    = (const float*)d_in[2];
    const float* F21    = (const float*)d_in[3];
    const float* Z1     = (const float*)d_in[4];
    const float* Z2     = (const float*)d_in[5];
    const float* feat11 = (const float*)d_in[6];
    const float* feat12 = (const float*)d_in[7];
    const float* feat13 = (const float*)d_in[8];
    const float* feat21 = (const float*)d_in[9];
    const float* feat22 = (const float*)d_in[10];
    const float* feat23 = (const float*)d_in[11];
    const float* merged = (const float*)d_in[12];
    const float* tptr   = (const float*)d_in[13];

    enum { D10 = 0, D11, D20, D21, HEAD, HEAD1, HEAD2, HEAD3, HEAD4,
           R01, R04, R05, R11, R14, R15, R21, R24, R25, U04, U05, U14, U15 };
    auto getblk = [&](int i) -> Blk {
        int base = 14 + i * 6;
        Blk b;
        b.w1 = (const float*)d_in[base + 0];
        b.b1 = (const float*)d_in[base + 1];
        b.w2 = (const float*)d_in[base + 2];
        b.b2 = (const float*)d_in[base + 3];
        b.a1 = (const float*)d_in[base + 4];
        b.a2 = (const float*)d_in[base + 5];
        return b;
    };
    const int TBASE = 14 + 22 * 6;  // 146
    const float* tw1 = (const float*)d_in[TBASE + 0];
    const float* tb1 = (const float*)d_in[TBASE + 1];
    const float* tw2 = (const float*)d_in[TBASE + 2];
    const float* tb2 = (const float*)d_in[TBASE + 3];
    const float* tw3 = (const float*)d_in[TBASE + 4];
    const float* tb3 = (const float*)d_in[TBASE + 5];
    const float* ta1 = (const float*)d_in[TBASE + 6];

    // Workspace bump allocator
    char* wsp = (char*)d_ws;
    size_t off = 0;
    auto alloc = [&](long nflt) -> float* {
        float* p = (float*)(wsp + off);
        off += (size_t)nflt * sizeof(float);
        off = (off + 255) & ~(size_t)255;
        return p;
    };

    float* F1t   = alloc(2 * PX);
    float* F2t   = alloc(2 * PX);
    float* F1td  = alloc(2 * PX2);
    float* F2td  = alloc(2 * PX2);
    float* Z1d   = alloc(PX2);
    float* Z2d   = alloc(PX2);
    float* F1tdd = alloc(2 * PX4);
    float* F2tdd = alloc(2 * PX4);
    float* Z1dd  = alloc(PX4);
    float* Z2dd  = alloc(PX4);
    float* xcat  = alloc(6 * PX);      // [I1t, I2t]
    float* x1cat = alloc(64 * PX);     // [f1t1, f2t1]
    float* x2cat = alloc(128 * PX2);   // [f1t2, f2t2]
    float* x3cat = alloc(192 * PX4);   // [f1t3, f2t3]
    float* num   = alloc(32 * PX);     // splat scratch (max C*hw over all calls)
    float* den   = alloc(PX);
    float* X00 = alloc(32 * PX);
    float* X01 = alloc(32 * PX);
    float* X04 = alloc(32 * PX);
    float* X05 = alloc(32 * PX);
    float* t32 = alloc(32 * PX);
    float* X10 = alloc(64 * PX2);
    float* X11 = alloc(64 * PX2);
    float* X14 = alloc(64 * PX2);
    float* X15 = alloc(64 * PX2);
    float* t64 = alloc(64 * PX2);
    float* X20 = alloc(96 * PX4);
    float* X21 = alloc(96 * PX4);
    float* X24 = alloc(96 * PX4);
    float* X25 = alloc(96 * PX4);
    float* t96 = alloc(96 * PX4);
    float* ubuf  = alloc(64 * PX);      // upsample scratch (covers 96*PX2 too)
    float* tb128 = alloc(128 * PX);     // tail conv2 output
    float* tshuf = alloc(128 * PX);     // pixel-shuffled, 32ch @ 896x1536
    _Float16* packbuf = (_Float16*)alloc(128 * 1024);  // 512KB packed-A scratch

    // ---- Stage 1: time-scaled flows -------------------------------------
    scale_flow<<<nblk(2 * PX, THREADS), THREADS, 0, stream>>>(F1t, F12, 2 * PX, tptr, 0.f, 1.f);
    scale_flow<<<nblk(2 * PX, THREADS), THREADS, 0, stream>>>(F2t, F21, 2 * PX, tptr, 1.f, -1.f);

    // ---- Stage 2: forward warps -----------------------------------------
    run_warp_norm(stream, I1,     3,  F1t, Z1, H, W, num, den, xcat);
    run_warp_norm(stream, I2,     3,  F2t, Z2, H, W, num, den, xcat + 3 * PX);
    run_warp_norm(stream, feat11, 32, F1t, Z1, H, W, num, den, x1cat);
    run_warp_norm(stream, feat21, 32, F2t, Z2, H, W, num, den, x1cat + 32 * PX);

    resize_bilinear_k<<<nblk(2 * PX2, THREADS), THREADS, 0, stream>>>(F1t, F1td, 2, H, W, H2, W2, 0.5f);
    resize_bilinear_k<<<nblk(PX2, THREADS), THREADS, 0, stream>>>(Z1, Z1d, 1, H, W, H2, W2, 1.f);
    resize_bilinear_k<<<nblk(2 * PX2, THREADS), THREADS, 0, stream>>>(F2t, F2td, 2, H, W, H2, W2, 0.5f);
    resize_bilinear_k<<<nblk(PX2, THREADS), THREADS, 0, stream>>>(Z2, Z2d, 1, H, W, H2, W2, 1.f);
    run_warp_norm(stream, feat12, 64, F1td, Z1d, H2, W2, num, den, x2cat);
    run_warp_norm(stream, feat22, 64, F2td, Z2d, H2, W2, num, den, x2cat + 64 * PX2);

    resize_bilinear_k<<<nblk(2 * PX4, THREADS), THREADS, 0, stream>>>(F1t, F1tdd, 2, H, W, H4, W4, 0.25f);
    resize_bilinear_k<<<nblk(PX4, THREADS), THREADS, 0, stream>>>(Z1, Z1dd, 1, H, W, H4, W4, 1.f);
    resize_bilinear_k<<<nblk(2 * PX4, THREADS), THREADS, 0, stream>>>(F2t, F2tdd, 2, H, W, H4, W4, 0.25f);
    resize_bilinear_k<<<nblk(PX4, THREADS), THREADS, 0, stream>>>(Z2, Z2dd, 1, H, W, H4, W4, 1.f);
    run_warp_norm(stream, feat13, 96, F1tdd, Z1dd, H4, W4, num, den, x3cat);
    run_warp_norm(stream, feat23, 96, F2tdd, Z2dd, H4, W4, num, den, x3cat + 96 * PX4);

    // ---- Stage 3: GridNet ----------------------------------------------
    run_res(stream, xcat,   6,   H, W, getblk(HEAD),  32, t32, X00, 0, packbuf);
    run_res(stream, x1cat,  64,  H, W, getblk(HEAD1), 32, t32, X00, 1, packbuf);
    run_res(stream, merged, 3,   H, W, getblk(HEAD4), 32, t32, X00, 1, packbuf);

    run_res(stream, X00, 32, H, W, getblk(R01), 32, t32, X01, 0, packbuf);
    add_f32<<<nblk(32 * PX, THREADS), THREADS, 0, stream>>>(X01, X00, 32 * PX);

    run_down(stream, X00, 32, H, W, getblk(D10), 64, t64, X10, 0, packbuf);
    run_res(stream, x2cat, 128, H2, W2, getblk(HEAD2), 64, t64, X10, 1, packbuf);

    run_down(stream, X10, 64, H2, W2, getblk(D20), 96, t96, X20, 0, packbuf);
    run_res(stream, x3cat, 192, H4, W4, getblk(HEAD3), 96, t96, X20, 1, packbuf);

    run_res(stream, X10, 64, H2, W2, getblk(R11), 64, t64, X11, 0, packbuf);
    add_f32<<<nblk(64 * PX2, THREADS), THREADS, 0, stream>>>(X11, X10, 64 * PX2);
    run_down(stream, X01, 32, H, W, getblk(D11), 64, t64, X11, 1, packbuf);

    run_res(stream, X20, 96, H4, W4, getblk(R21), 96, t96, X21, 0, packbuf);
    add_f32<<<nblk(96 * PX4, THREADS), THREADS, 0, stream>>>(X21, X20, 96 * PX4);
    run_down(stream, X11, 64, H2, W2, getblk(D21), 96, t96, X21, 1, packbuf);

    run_res(stream, X21, 96, H4, W4, getblk(R24), 96, t96, X24, 0, packbuf);
    add_f32<<<nblk(96 * PX4, THREADS), THREADS, 0, stream>>>(X24, X21, 96 * PX4);

    run_res(stream, X24, 96, H4, W4, getblk(R25), 96, t96, X25, 0, packbuf);
    add_f32<<<nblk(96 * PX4, THREADS), THREADS, 0, stream>>>(X25, X24, 96 * PX4);

    run_up(stream, X24, 96, H4, W4, getblk(U14), 64, ubuf, t64, X14, 0, packbuf);
    run_res(stream, X11, 64, H2, W2, getblk(R14), 64, t64, X14, 1, packbuf);
    add_f32<<<nblk(64 * PX2, THREADS), THREADS, 0, stream>>>(X14, X11, 64 * PX2);

    run_up(stream, X14, 64, H2, W2, getblk(U04), 32, ubuf, t32, X04, 0, packbuf);
    run_res(stream, X01, 32, H, W, getblk(R04), 32, t32, X04, 1, packbuf);
    add_f32<<<nblk(32 * PX, THREADS), THREADS, 0, stream>>>(X04, X01, 32 * PX);

    run_up(stream, X25, 96, H4, W4, getblk(U15), 64, ubuf, t64, X15, 0, packbuf);
    run_res(stream, X14, 64, H2, W2, getblk(R15), 64, t64, X15, 1, packbuf);
    add_f32<<<nblk(64 * PX2, THREADS), THREADS, 0, stream>>>(X15, X14, 64 * PX2);

    run_up(stream, X15, 64, H2, W2, getblk(U05), 32, ubuf, t32, X05, 0, packbuf);
    run_res(stream, X04, 32, H, W, getblk(R05), 32, t32, X05, 1, packbuf);
    add_f32<<<nblk(32 * PX, THREADS), THREADS, 0, stream>>>(X05, X04, 32 * PX);

    // ---- Stage 4: tail --------------------------------------------------
    launch_conv(stream, X05, 32, H, W, tw1, tb1, nullptr, t32, 32, 1, 0, 0, packbuf);
    launch_conv(stream, t32, 32, H, W, tw2, tb2, ta1, tb128, 128, 1, 0, 0, packbuf);
    pixel_shuffle2<<<nblk(128 * PX, THREADS), THREADS, 0, stream>>>(tb128, tshuf, 32, H, W);
    launch_conv(stream, tshuf, 32, 2 * H, 2 * W, tw3, tb3, nullptr, (float*)d_out, 3, 1, 0, 1, packbuf);
}
